// Graph_Model_40467181863338
// MI455X (gfx1250) — matrix operand order repaired and checked by reading, not compiled, not run on hardware
//
#include <hip/hip_runtime.h>

// ---------------------------------------------------------------------------
// MI455X (gfx1250) implementation.
// Dominant compute: adj[4096x4096] @ f[b][4096xL] (~94 GFLOP) + feature GEMMs
// (~10 GFLOP). adj as bf16 (33MB) is L2-resident (192MB L2), re-read 64x, so
// we are matrix-pipe bound -> v_wmma_f32_16x16x32_bf16 with f32 accumulation.
// GEMM inner loops: K unrolled by 2 with statically-named double buffers
// (a0/b0[NT] vs a1/b1[NT]) so ~10 b128 loads fly behind each WMMA chain, and
// ALL addressing reduced to two running per-lane pointers + constant
// instruction offsets (tile strides are compile-time) -> no pointer spills,
// minimal address VALU in the hot loop.
// ---------------------------------------------------------------------------

#define PI_F 3.14159265358979323846f

typedef __attribute__((ext_vector_type(16))) __bf16 v16bf;
typedef __attribute__((ext_vector_type(8)))  float  v8f;

union AFrag { v16bf v; uint4 u[2]; };
union Pack8 { __bf16 h[8]; uint4 u; };
union Pack4 { __bf16 h[4]; uint2 u; };

__device__ __forceinline__ v8f wmma_bf16(v16bf a, v16bf b, v8f c) {
  // 8 args: (neg_a, A, neg_b, B, c_mod, C, reuse_a, reuse_b)
  return __builtin_amdgcn_wmma_f32_16x16x32_bf16(false, a, false, b, (short)0, c,
                                                 false, false);
}

// --------------------------- prep kernels ----------------------------------

// fp32 -> bf16, 4 at a time (adj: 16.7M elements)
__global__ void cvt_bf16_vec(const float4* __restrict__ src,
                             __bf16* __restrict__ dst, int n4) {
  int i = blockIdx.x * blockDim.x + threadIdx.x;
  if (i >= n4) return;
  float4 v = src[i];
  Pack4 p;
  p.h[0] = (__bf16)v.x; p.h[1] = (__bf16)v.y;
  p.h[2] = (__bf16)v.z; p.h[3] = (__bf16)v.w;
  *(uint2*)(dst + (size_t)i * 4) = p.u;
}

// gw [K][N] fp32 row-major -> gwT [Npad][Kpad] bf16 (zero padded)
__global__ void make_gwT(const float* __restrict__ gw, __bf16* __restrict__ gwT,
                         int K, int Kpad, int N, int Npad) {
  int i = blockIdx.x * blockDim.x + threadIdx.x;
  if (i >= Npad * Kpad) return;
  int n = i / Kpad, k = i % Kpad;
  float v = (n < N && k < K) ? gw[(size_t)k * N + n] : 0.f;
  gwT[i] = (__bf16)v;
}

// --------------------------- action MLP ------------------------------------
// mask[16,50] -> 200 -> 100 -> 100 ; output bf16 [16][100]. One block.
__global__ void action_mlp(const float* __restrict__ mask,
                           const float* __restrict__ aw1, const float* __restrict__ ab1,
                           const float* __restrict__ aw2, const float* __restrict__ ab2,
                           const float* __restrict__ aw3, const float* __restrict__ ab3,
                           __bf16* __restrict__ a_out) {
  __shared__ float sm[16 * 50];
  __shared__ float h1[16 * 200];
  __shared__ float h2[16 * 100];
  int tid = threadIdx.x;
  for (int i = tid; i < 16 * 50; i += 256) sm[i] = mask[i];
  __syncthreads();
  for (int i = tid; i < 16 * 200; i += 256) {
    int r = i / 200, c = i % 200;
    float acc = ab1[c];
    for (int k = 0; k < 50; k++) acc += sm[r * 50 + k] * aw1[k * 200 + c];
    h1[i] = fmaxf(acc, 0.f);
  }
  __syncthreads();
  for (int i = tid; i < 16 * 100; i += 256) {
    int r = i / 100, c = i % 100;
    float acc = ab2[c];
    for (int k = 0; k < 200; k++) acc += h1[r * 200 + k] * aw2[k * 100 + c];
    h2[i] = fmaxf(acc, 0.f);
  }
  __syncthreads();
  for (int i = tid; i < 16 * 100; i += 256) {
    int r = i / 100, c = i % 100;
    float acc = ab3[c];
    for (int k = 0; k < 100; k++) acc += h2[r * 100 + k] * aw3[k * 100 + c];
    a_out[i] = (__bf16)acc;   // no relu on last action layer
  }
}

// ------------------- positional MLP + x0 assembly --------------------------
// Per point: nerf(mesh)->63 feats, MLP 63->25->50->100; x0[b][n][0:100]=a,
// [100:200]=p, [200:300]=emb[mask_idx], [300:320]=0.  x0 bf16 stride 320.
__global__ void pos_assemble(const float* __restrict__ mesh,
                             const int* __restrict__ mask_idx,
                             const float* __restrict__ emb,
                             const float* __restrict__ pw1, const float* __restrict__ pb1,
                             const float* __restrict__ pw2, const float* __restrict__ pb2,
                             const float* __restrict__ pw3, const float* __restrict__ pb3,
                             const __bf16* __restrict__ a_bf,
                             __bf16* __restrict__ x0) {
  // LDS weight staging: pw1@0(1575) pb1@1575(25) pw2@1600(1250) pb2@2850(50)
  //                     pw3@2900(5000) pb3@7900(100)  => 8000 floats (32KB)
  __shared__ float W[8000];
  int tid = threadIdx.x;
  for (int i = tid; i < 1575; i += 256) W[i] = pw1[i];
  for (int i = tid; i < 25;   i += 256) W[1575 + i] = pb1[i];
  for (int i = tid; i < 1250; i += 256) W[1600 + i] = pw2[i];
  for (int i = tid; i < 50;   i += 256) W[2850 + i] = pb2[i];
  for (int i = tid; i < 5000; i += 256) W[2900 + i] = pw3[i];
  for (int i = tid; i < 100;  i += 256) W[7900 + i] = pb3[i];
  __syncthreads();

  int pt = blockIdx.x * blockDim.x + tid;   // 0..65535
  int b = pt >> 12;
  float pd[3];
  pd[0] = mesh[(size_t)pt * 3 + 0];
  pd[1] = mesh[(size_t)pt * 3 + 1];
  pd[2] = mesh[(size_t)pt * 3 + 2];

  // layer 1: 63 -> 25 (k-outer so h1 stays in registers)
  float h1[25];
#pragma unroll
  for (int c = 0; c < 25; c++) h1[c] = W[1575 + c];
  for (int k = 0; k < 63; k++) {
    float xv;
    if (k < 60) {
      int f = k / 6, r = k % 6, d = r % 3;
      float freq = (f == 0) ? PI_F : 2.f * PI_F * (float)f;
      float ang = pd[d] * freq;
      xv = (r < 3) ? __sinf(ang) : __cosf(ang);
    } else {
      xv = pd[k - 60];
    }
#pragma unroll
    for (int c = 0; c < 25; c++) h1[c] += xv * W[k * 25 + c];
  }
#pragma unroll
  for (int c = 0; c < 25; c++) h1[c] = fmaxf(h1[c], 0.f);

  // layer 2: 25 -> 50
  float h2[50];
#pragma unroll
  for (int c = 0; c < 50; c++) h2[c] = W[2850 + c];
  for (int k = 0; k < 25; k++) {
    float xv = h1[k];
#pragma unroll
    for (int c = 0; c < 50; c++) h2[c] += xv * W[1600 + k * 50 + c];
  }
#pragma unroll
  for (int c = 0; c < 50; c++) h2[c] = fmaxf(h2[c], 0.f);

  __bf16* xr = x0 + (size_t)pt * 320;

  // layer 3: 50 -> 100 (two halves of 50 to bound VGPR pressure), no relu
  for (int hh = 0; hh < 2; hh++) {
    float p3[50];
#pragma unroll
    for (int c = 0; c < 50; c++) p3[c] = W[7900 + hh * 50 + c];
    for (int k = 0; k < 50; k++) {
      float xv = h2[k];
#pragma unroll
      for (int c = 0; c < 50; c++) p3[c] += xv * W[2900 + k * 100 + hh * 50 + c];
    }
#pragma unroll
    for (int c = 0; c < 50; c++) xr[100 + hh * 50 + c] = (__bf16)p3[c];
  }

  // action broadcast + mask embedding + pad
  for (int c = 0; c < 100; c++) xr[c] = a_bf[b * 100 + c];
  int mi = mask_idx[pt];
  for (int c = 0; c < 100; c++) xr[200 + c] = (__bf16)emb[mi * 100 + c];
  for (int c = 0; c < 20; c++)  xr[300 + c] = (__bf16)0.f;
}

// --------------------------- feature GEMM ----------------------------------
// f = x @ gw : x bf16 [16][4096][KPAD] (row-major), gwT bf16 [Nout][KPAD].
// blockIdx.x: 512 = 16 batches x 32 row-blocks of 128; blockIdx.y: column
// half (4 tiles = 64 cols each).  KPAD is a template constant so the per-tile
// B offsets (t*16*KPAD) fold into the load-instruction immediates; only two
// running per-lane pointers (pa, pb) live in the loop.
template <bool WRIGHT, int KPAD>
__global__ void feat_gemm4(const __bf16* __restrict__ x, const __bf16* __restrict__ gwT,
                           __bf16* __restrict__ fT, __bf16* __restrict__ xnext,
                           int L, int FTT) {
  int blk = blockIdx.x;
  int b = blk >> 5, rb = blk & 31;
  int toff = blockIdx.y * 4;                  // global column-tile offset
  int tid = threadIdx.x;
  int wave = tid >> 5, lane = tid & 31;
  int half = lane >> 4, ln = lane & 15;
  int row = rb * 128 + wave * 16 + ln;
  const __bf16* pa = x + ((size_t)b * 4096 + row) * KPAD + half * 8;
  const __bf16* pb = gwT + ((size_t)(toff * 16 + ln)) * KPAD + half * 8;
  constexpr int TB = 16 * KPAD;               // B tile stride (elements)
  v8f acc[4] = {};

  AFrag a0, a1;
  AFrag b0[4], b1[4];

  auto loadA = [&](AFrag& a, const __bf16* p, int off) {
    a.u[0] = *(const uint4*)(p + off);
    a.u[1] = *(const uint4*)(p + off + 16);
  };
  auto loadB = [&](AFrag (&bf)[4], const __bf16* p, int off) {
#pragma unroll
    for (int t = 0; t < 4; t++) {
      bf[t].u[0] = *(const uint4*)(p + t * TB + off);
      bf[t].u[1] = *(const uint4*)(p + t * TB + off + 16);
    }
  };

  loadA(a0, pa, 0);
  loadB(b0, pb, 0);
  for (int k = 0; k < KPAD; k += 64) {
    loadA(a1, pa, 32);                        // in flight during chain 0
    loadB(b1, pb, 32);
#pragma unroll
    for (int t = 0; t < 4; t++) acc[t] = wmma_bf16(a0.v, b0[t].v, acc[t]);
    pa += 64; pb += 64;
    if (k + 64 < KPAD) {                      // in flight during chain 1
      loadA(a0, pa, 0);
      loadB(b0, pb, 0);
    }
#pragma unroll
    for (int t = 0; t < 4; t++) acc[t] = wmma_bf16(a1.v, b1[t].v, acc[t]);
  }

  int m0 = rb * 128 + wave * 16 + half * 8;   // C layout: m = m0+v, n = tg*16+ln
#pragma unroll
  for (int t = 0; t < 4; t++) {
    int tg = toff + t;
    if (tg < FTT) {                           // transposed store: 16B contiguous
      Pack8 p;
#pragma unroll
      for (int v = 0; v < 8; v++) p.h[v] = (__bf16)acc[t][v];
      *(uint4*)(fT + ((size_t)b * 64 + tg * 16 + ln) * 4096 + m0) = p.u;
    }
    if constexpr (WRIGHT) {
      int c = tg * 16 + ln;
      if (c >= L) {
#pragma unroll
        for (int v = 0; v < 8; v++) {
          float val = acc[t][v];
          val = val > 0.f ? val : 0.f;
          xnext[((size_t)b * 4096 + m0 + v) * 128 + c] = (__bf16)val;
        }
      }
    }
  }
}

// ------------------------- aggregation GEMM --------------------------------
// y[b] = adj @ f[b] : A = adj bf16 [4096][4096] (L2-resident), B = fT[b][c][j]
// (row stride 4096 -> tile stride t*131072 B folds into load immediates).
// Two running per-lane pointers, fully double-buffered K loop (unroll 2).
// Non-last: x_next[...,c<L] = relu(y + bias[c]).  Last: ylast fp32 [b][i][64].
template <int NT, bool LAST>
__global__ void agg_gemm(const __bf16* __restrict__ adjb, const __bf16* __restrict__ fT,
                         const float* __restrict__ bias, __bf16* __restrict__ xnext,
                         float* __restrict__ ylast, int L) {
  int blk = blockIdx.x;
  int b = blk >> 5, rb = blk & 31;
  int tid = threadIdx.x;
  int wave = tid >> 5, lane = tid & 31;
  int half = lane >> 4, ln = lane & 15;
  int row = rb * 128 + wave * 16 + ln;
  const __bf16* pa = adjb + (size_t)row * 4096 + half * 8;
  const __bf16* pb = fT + ((size_t)b * 64 + ln) * 4096 + half * 8;
  constexpr int TB = 16 * 4096;               // B tile stride (elements)
  v8f acc[NT] = {};

  AFrag a0, a1;
  AFrag b0[NT], b1[NT];

  auto loadA = [&](AFrag& a, const __bf16* p, int off) {
    a.u[0] = *(const uint4*)(p + off);
    a.u[1] = *(const uint4*)(p + off + 16);
  };
  auto loadB = [&](AFrag (&bf)[NT], const __bf16* p, int off) {
#pragma unroll
    for (int t = 0; t < NT; t++) {
      bf[t].u[0] = *(const uint4*)(p + t * TB + off);
      bf[t].u[1] = *(const uint4*)(p + t * TB + off + 16);
    }
  };

  loadA(a0, pa, 0);
  loadB(b0, pb, 0);
  for (int k = 0; k < 4096; k += 64) {
    loadA(a1, pa, 32);                        // in flight during chain 0
    loadB(b1, pb, 32);
    __builtin_prefetch(pa + 256, 0, 1);       // pull adj ahead into caches
#pragma unroll
    for (int t = 0; t < NT; t++) acc[t] = wmma_bf16(a0.v, b0[t].v, acc[t]);
    pa += 64; pb += 64;
    if (k + 64 < 4096) {                      // in flight during chain 1
      loadA(a0, pa, 0);
      loadB(b0, pb, 0);
    }
#pragma unroll
    for (int t = 0; t < NT; t++) acc[t] = wmma_bf16(a1.v, b1[t].v, acc[t]);
  }

  int m0 = rb * 128 + wave * 16 + half * 8;
  if constexpr (!LAST) {
#pragma unroll
    for (int t = 0; t < NT; t++) {
      int c = t * 16 + ln;
      if (c < L) {
        float bb = bias[c];
#pragma unroll
        for (int v = 0; v < 8; v++) {
          float val = acc[t][v] + bb;
          val = val > 0.f ? val : 0.f;
          xnext[((size_t)b * 4096 + m0 + v) * 128 + c] = (__bf16)val;
        }
      }
    }
  } else {
#pragma unroll
    for (int t = 0; t < NT; t++) {
      int c = t * 16 + ln;
      float bb = (c < 50) ? bias[c] : 0.f;
#pragma unroll
      for (int v = 0; v < 8; v++)
        ylast[((size_t)b * 4096 + m0 + v) * 64 + c] = acc[t][v] + bb;
    }
  }
}

// ------------------------------ max reduce ---------------------------------
// out[b][c] = max_i ylast[b][i][c].  64 blocks: b = blk>>2, col-tile = blk&3.
__global__ void reduce_max(const float* __restrict__ y, float* __restrict__ out) {
  int b = blockIdx.x >> 2, ct = blockIdx.x & 3;
  int tid = threadIdx.x;
  int cl = tid & 15, rg = tid >> 4;           // 16 cols x 16 row-groups
  int c = ct * 16 + cl;
  float m = -3.402823466e38f;
  for (int i = rg; i < 4096; i += 16)
    m = fmaxf(m, y[((size_t)b * 4096 + i) * 64 + c]);
  __shared__ float s[256];
  s[tid] = m;
  __syncthreads();
  for (int step = 8; step >= 1; step >>= 1) {
    if (rg < step) s[tid] = fmaxf(s[tid], s[tid + step * 16]);
    __syncthreads();
  }
  if (rg == 0 && c < 50) out[b * 50 + c] = s[tid];
}

// ------------------------------ launcher -----------------------------------
extern "C" void kernel_launch(void* const* d_in, const int* in_sizes, int n_in,
                              void* d_out, int out_size, void* d_ws, size_t ws_size,
                              hipStream_t stream) {
  const float* mask     = (const float*)d_in[0];
  const float* mesh     = (const float*)d_in[1];
  const int*   mask_idx = (const int*)d_in[2];
  const float* adj      = (const float*)d_in[3];
  const float* aw1 = (const float*)d_in[4];  const float* ab1 = (const float*)d_in[5];
  const float* aw2 = (const float*)d_in[6];  const float* ab2 = (const float*)d_in[7];
  const float* aw3 = (const float*)d_in[8];  const float* ab3 = (const float*)d_in[9];
  const float* pw1 = (const float*)d_in[10]; const float* pb1 = (const float*)d_in[11];
  const float* pw2 = (const float*)d_in[12]; const float* pb2 = (const float*)d_in[13];
  const float* pw3 = (const float*)d_in[14]; const float* pb3 = (const float*)d_in[15];
  const float* emb = (const float*)d_in[16];
  const float* gw0 = (const float*)d_in[17]; const float* gb0 = (const float*)d_in[18];
  const float* gw1 = (const float*)d_in[19]; const float* gb1 = (const float*)d_in[20];
  const float* gw2 = (const float*)d_in[21]; const float* gb2 = (const float*)d_in[22];
  const float* gw3 = (const float*)d_in[23]; const float* gb3 = (const float*)d_in[24];
  float* out = (float*)d_out;

  // workspace suballocation (~101 MB); x0 / xB / ylast are time-aliased.
  char* ws = (char*)d_ws;
  size_t off = 0;
  auto alloc = [&](size_t bytes) {
    size_t p = off;
    off += (bytes + 255) & ~(size_t)255;
    return p;
  };
  __bf16* adjb = (__bf16*)(ws + alloc(4096ull * 4096 * 2));      // 33.6 MB
  __bf16* gwT0 = (__bf16*)(ws + alloc(128 * 320 * 2));
  __bf16* gwT1 = (__bf16*)(ws + alloc(128 * 128 * 2));
  __bf16* gwT2 = (__bf16*)(ws + alloc(128 * 128 * 2));
  __bf16* gwT3 = (__bf16*)(ws + alloc(64 * 128 * 2));
  __bf16* a_bf = (__bf16*)(ws + alloc(16 * 100 * 2));
  __bf16* fT   = (__bf16*)(ws + alloc(16ull * 64 * 4096 * 2));   // 8.4 MB
  __bf16* xA   = (__bf16*)(ws + alloc(16ull * 4096 * 128 * 2));  // 16.8 MB
  char*   big  = ws + alloc(16ull * 4096 * 320 * 2);             // 41.9 MB
  __bf16* x0   = (__bf16*)big;        // live: pos_assemble .. feat layer0
  __bf16* xB   = (__bf16*)big;        // live: layer1 .. feat layer2
  float*  ylast = (float*)big;        // live: agg layer3 .. reduce

  // prep
  cvt_bf16_vec<<<(4096 * 4096 / 4 + 255) / 256, 256, 0, stream>>>(
      (const float4*)adj, adjb, 4096 * 4096 / 4);
  make_gwT<<<(128 * 320 + 255) / 256, 256, 0, stream>>>(gw0, gwT0, 300, 320, 128, 128);
  make_gwT<<<(128 * 128 + 255) / 256, 256, 0, stream>>>(gw1, gwT1, 128, 128, 128, 128);
  make_gwT<<<(128 * 128 + 255) / 256, 256, 0, stream>>>(gw2, gwT2, 128, 128, 128, 128);
  make_gwT<<<(64 * 128 + 255) / 256, 256, 0, stream>>>(gw3, gwT3, 128, 128, 50, 64);

  // front end
  action_mlp<<<1, 256, 0, stream>>>(mask, aw1, ab1, aw2, ab2, aw3, ab3, a_bf);
  pos_assemble<<<256, 256, 0, stream>>>(mesh, mask_idx, emb, pw1, pb1, pw2, pb2,
                                        pw3, pb3, a_bf, x0);

  // GCN layers (x: 512 blocks = 16 batches x 32 row-blocks; y: column halves)
  feat_gemm4<true, 320><<<dim3(512, 2), 256, 0, stream>>>(x0, gwT0, fT, xA, 42, 3);
  agg_gemm<3, false><<<512, 256, 0, stream>>>(adjb, fT, gb0, xA, nullptr, 42);

  feat_gemm4<true, 128><<<dim3(512, 2), 256, 0, stream>>>(xA, gwT1, fT, xB, 42, 3);
  agg_gemm<3, false><<<512, 256, 0, stream>>>(adjb, fT, gb1, xB, nullptr, 42);

  feat_gemm4<true, 128><<<dim3(512, 2), 256, 0, stream>>>(xB, gwT2, fT, xA, 42, 3);
  agg_gemm<3, false><<<512, 256, 0, stream>>>(adjb, fT, gb2, xA, nullptr, 42);

  feat_gemm4<false, 128><<<dim3(512, 1), 256, 0, stream>>>(xA, gwT3, fT, nullptr, 42, 4);
  agg_gemm<4, true><<<512, 256, 0, stream>>>(adjb, fT, gb3, nullptr, ylast, 42);

  reduce_max<<<64, 256, 0, stream>>>(ylast, out);
}
